// Attention_27685359190353
// MI455X (gfx1250) — matrix operand order repaired and checked by reading, not compiled, not run on hardware
//
#include <hip/hip_runtime.h>
#include <hip/hip_bf16.h>

#define SEQ 2048
#define DIM 2048
#define NH 32
#define NKV 8
#define HD 64
#define HALF 32
#define KVD 512

typedef __attribute__((ext_vector_type(16))) __bf16 v16bf;
typedef __attribute__((ext_vector_type(8)))  float  v8f;

union FragBF { v16bf v; unsigned int u[8]; };
union AccF   { v8f   v; float f[8]; };

__device__ __forceinline__ unsigned short f2bf(float x) {
    __hip_bfloat16 h = __float2bfloat16(x);
    return __builtin_bit_cast(unsigned short, h);
}

// --- CDNA5 async global->LDS copy (ASYNCcnt-tracked), per-lane 16B ----------
__device__ __forceinline__ unsigned lds_off(const void* p) {
    return (unsigned)(unsigned long long)(uintptr_t)p;   // low 32 bits of generic LDS ptr = LDS offset
}
__device__ __forceinline__ void async_b128(unsigned ldsDst, const unsigned short* src) {
    asm volatile("global_load_async_to_lds_b128 %0, %1, off"
                 :: "v"(ldsDst), "v"((unsigned long long)(uintptr_t)src)
                 : "memory");
}
__device__ __forceinline__ void wait_async() {
    asm volatile("s_wait_asynccnt 0x0" ::: "memory");
}
__device__ __forceinline__ void wait_ds() {
    asm volatile("s_wait_dscnt 0x0" ::: "memory");
}

// ---------------------------------------------------------------------------
// fp32 -> bf16 bulk convert (x)
// ---------------------------------------------------------------------------
__global__ __launch_bounds__(256) void k_f32_to_bf16(const float* __restrict__ s,
                                                     unsigned short* __restrict__ d, int n) {
    for (int i = blockIdx.x * blockDim.x + threadIdx.x; i < n; i += gridDim.x * blockDim.x)
        d[i] = f2bf(s[i]);
}

// fp32 [K=2048][N] -> bf16 transposed [N][K]  (weights, done once)
__global__ __launch_bounds__(256) void k_convT_bf16(const float* __restrict__ s,
                                                    unsigned short* __restrict__ d, int N) {
    int idx = blockIdx.x * blockDim.x + threadIdx.x;
    if (idx >= N * DIM) return;
    int k = idx & (DIM - 1);
    int n = idx >> 11;
    d[(size_t)n * DIM + k] = f2bf(s[(size_t)k * N + n]);
}

// ---------------------------------------------------------------------------
// bf16 WMMA GEMM: C[M,N] f32 = A[M,K] @ B[K,N], with B passed TRANSPOSED
// (BT[N][K] bf16).  Block = 256 threads (8 waves, 2x4), block tile 64x256,
// wave tile 32x64 (2x4 WMMA -> 8 wmma per K-step, A frags reused x4, B x2).
// Double-buffered LDS; tiles staged with global_load_async_to_lds_b128 so
// WMMA overlaps the next tile's global traffic.
// ---------------------------------------------------------------------------
__global__ __launch_bounds__(256) void k_gemm_bf16(const unsigned short* __restrict__ A,
                                                   const unsigned short* __restrict__ BT,
                                                   float* __restrict__ C,
                                                   int M, int N, int K) {
    __shared__ __align__(16) unsigned short at[2][64 * 32];    // [row][k]
    __shared__ __align__(16) unsigned short bt[2][256 * 32];   // [n][k]
    const int tid  = threadIdx.x;
    const int lane = tid & 31;
    const int wave = tid >> 5;
    const int hl   = lane >> 4;
    const int l15  = lane & 15;
    const int m0 = blockIdx.y * 64;
    const int n0 = blockIdx.x * 256;
    const int wr = (wave >> 2) * 32;
    const int wc = (wave & 3)  * 64;

    const int arow = tid >> 2, aseg = tid & 3;   // A tile: 256 x b128

    auto stage = [&](int buf, int k0) {
        async_b128(lds_off(&at[buf][arow * 32 + aseg * 8]),
                   A + (size_t)(m0 + arow) * K + k0 + aseg * 8);
        #pragma unroll
        for (int t = 0; t < 4; t++) {            // B tile: 1024 x b128
            int idx = tid + t * 256;
            int n = idx >> 2, seg = idx & 3;
            async_b128(lds_off(&bt[buf][n * 32 + seg * 8]),
                       BT + (size_t)(n0 + n) * K + k0 + seg * 8);
        }
    };

    AccF acc[2][4];
    #pragma unroll
    for (int i = 0; i < 2; i++)
        #pragma unroll
        for (int j = 0; j < 4; j++)
            #pragma unroll
            for (int r = 0; r < 8; r++) acc[i][j].f[r] = 0.f;

    stage(0, 0);
    const int iters = K / 32;
    for (int it = 0; it < iters; it++) {
        wait_async();
        __syncthreads();                          // all waves' async copies landed
        if (it + 1 < iters) stage((it + 1) & 1, (it + 1) * 32);

        const unsigned short* atc = at[it & 1];
        const unsigned short* btc = bt[it & 1];
        FragBF afr[2], bfr[4];
        #pragma unroll
        for (int i2 = 0; i2 < 2; i2++) {
            int row = wr + i2 * 16 + l15;
            #pragma unroll
            for (int i = 0; i < 8; i++) {
                int k = 2 * (i & 3) + ((i & 4) ? 16 : 0) + 8 * hl;   // A 16-bit layout
                afr[i2].u[i] = *(const unsigned int*)&atc[row * 32 + k];
            }
        }
        #pragma unroll
        for (int j2 = 0; j2 < 4; j2++) {
            int col = wc + j2 * 16 + l15;
            #pragma unroll
            for (int i = 0; i < 8; i++) {
                int k = 2 * (i & 3) + ((i & 4) ? 16 : 0) + 8 * hl;   // B 16-bit layout
                bfr[j2].u[i] = *(const unsigned int*)&btc[col * 32 + k];
            }
        }
        #pragma unroll
        for (int i2 = 0; i2 < 2; i2++)
            #pragma unroll
            for (int j2 = 0; j2 < 4; j2++)
                acc[i2][j2].v = __builtin_amdgcn_wmma_f32_16x16x32_bf16(
                    false, afr[i2].v, false, bfr[j2].v, (short)0, acc[i2][j2].v, false, false);
        // no trailing barrier: next iteration's wait_async + barrier orders
        // the buffer reuse (this iter's ds reads complete before the wmma).
    }

    #pragma unroll
    for (int i2 = 0; i2 < 2; i2++)
        #pragma unroll
        for (int j2 = 0; j2 < 4; j2++)
            #pragma unroll
            for (int r = 0; r < 8; r++) {
                int row = m0 + wr + i2 * 16 + r + 8 * hl;
                int col = n0 + wc + j2 * 16 + l15;
                C[(size_t)row * N + col] = acc[i2][j2].f[r];
            }
}

// ---------------------------------------------------------------------------
// RoPE + repack to head-major bf16: src [SEQ][nh*64] f32 -> dst [nh][SEQ][64]
// ---------------------------------------------------------------------------
__global__ __launch_bounds__(256) void k_rope_pack(const float* __restrict__ src,
                                                   unsigned short* __restrict__ dst,
                                                   const float* __restrict__ cosb,
                                                   const float* __restrict__ sinb, int nh) {
    int idx = blockIdx.x * blockDim.x + threadIdx.x;
    if (idx >= SEQ * nh * HALF) return;
    int i  = idx & 31;
    int t  = idx >> 5;
    int hh = t % nh;
    int s  = t / nh;
    float c  = cosb[s * HALF + i];
    float sn = sinb[s * HALF + i];
    float xr = src[(size_t)s * (nh * HD) + hh * HD + 2 * i];
    float xi = src[(size_t)s * (nh * HD) + hh * HD + 2 * i + 1];
    size_t ob = (size_t)hh * SEQ * HD + (size_t)s * HD + 2 * i;
    dst[ob]     = f2bf(xr * c - xi * sn);
    dst[ob + 1] = f2bf(xr * sn + xi * c);
}

// V repack: [SEQ][512] f32 -> TRANSPOSED head-major [NKV][64][SEQ] bf16
__global__ __launch_bounds__(256) void k_v_packT(const float* __restrict__ src,
                                                 unsigned short* __restrict__ dst) {
    int idx = blockIdx.x * blockDim.x + threadIdx.x;
    if (idx >= SEQ * KVD) return;
    int s  = idx & (SEQ - 1);
    int t  = idx >> 11;
    int d  = t & 63;
    int hh = t >> 6;
    dst[((size_t)hh * HD + d) * SEQ + s] = f2bf(src[(size_t)s * KVD + hh * HD + d]);
}

// ---------------------------------------------------------------------------
// Flash attention: grid(32 qblocks, 32 heads), block = 128 threads (4 waves).
// Double-buffered K [key][d] and V [d][key] tiles staged by async-to-LDS
// (both contiguous copies thanks to the pre-transposed V); prefetch of the
// next key tile overlaps S/softmax/PV of the current one.
// ---------------------------------------------------------------------------
__global__ __launch_bounds__(128) void k_flash_attn(const unsigned short* __restrict__ Qh,
                                                    const unsigned short* __restrict__ Kh,
                                                    const unsigned short* __restrict__ VhT,
                                                    unsigned short* __restrict__ Ob) {
    __shared__ __align__(16) unsigned short kt[2][32 * 64];  // [key][d]
    __shared__ __align__(16) unsigned short vt[2][64 * 32];  // [d][key]
    __shared__ __align__(16) unsigned short pb[4][16 * 32];  // per-wave P tile

    const int tid  = threadIdx.x;
    const int lane = tid & 31;
    const int wave = tid >> 5;
    const int hl   = lane >> 4;
    const int l15  = lane & 15;
    const int h    = blockIdx.y;
    const int qb   = blockIdx.x;
    const int kvh  = h >> 2;
    const unsigned short* Q   = Qh  + (size_t)h   * SEQ * HD;
    const unsigned short* Kp  = Kh  + (size_t)kvh * SEQ * HD;
    const unsigned short* VpT = VhT + (size_t)kvh * HD * SEQ;
    const int qbase = qb * 64 + wave * 16;

    auto stage = [&](int buf, int k0) {
        #pragma unroll
        for (int t = 0; t < 2; t++) {            // K tile: 2048 bf16 contiguous
            int idx = tid + t * 128;
            async_b128(lds_off(&kt[buf][idx * 8]), Kp + (size_t)k0 * HD + idx * 8);
        }
        #pragma unroll
        for (int t = 0; t < 2; t++) {            // V tile: 64 rows x 64B
            int idx = tid + t * 128;
            int d = idx >> 2, seg = idx & 3;
            async_b128(lds_off(&vt[buf][d * 32 + seg * 8]),
                       VpT + (size_t)d * SEQ + k0 + seg * 8);
        }
    };

    FragBF qf[2];
    {
        int row = qbase + l15;
        #pragma unroll
        for (int d2 = 0; d2 < 2; d2++)
            #pragma unroll
            for (int i = 0; i < 8; i++) {
                int k = d2 * 32 + 2 * (i & 3) + ((i & 4) ? 16 : 0) + 8 * hl;
                qf[d2].u[i] = *(const unsigned int*)&Q[(size_t)row * HD + k];
            }
    }

    AccF o[4];
    float mrow[8], lrow[8];
    #pragma unroll
    for (int t = 0; t < 4; t++)
        #pragma unroll
        for (int r = 0; r < 8; r++) o[t].f[r] = 0.f;
    #pragma unroll
    for (int r = 0; r < 8; r++) { mrow[r] = -1e30f; lrow[r] = 0.f; }

    const int nkt = (qb + 1) * 2;
    stage(0, 0);
    for (int kti = 0; kti < nkt; kti++) {
        const int k0 = kti * 32;
        wait_async();
        __syncthreads();
        if (kti + 1 < nkt) stage((kti + 1) & 1, (kti + 1) * 32);
        const unsigned short* ktc = kt[kti & 1];
        const unsigned short* vtc = vt[kti & 1];

        // S = Q @ K^T (16x32)
        AccF s[2];
        #pragma unroll
        for (int j = 0; j < 2; j++) {
            #pragma unroll
            for (int r = 0; r < 8; r++) s[j].f[r] = 0.f;
            int keyloc = j * 16 + l15;
            #pragma unroll
            for (int d2 = 0; d2 < 2; d2++) {
                FragBF kf;
                #pragma unroll
                for (int i = 0; i < 8; i++) {
                    int d = d2 * 32 + 2 * (i & 3) + ((i & 4) ? 16 : 0) + 8 * hl;
                    kf.u[i] = *(const unsigned int*)&ktc[keyloc * 64 + d];
                }
                s[j].v = __builtin_amdgcn_wmma_f32_16x16x32_bf16(
                    false, qf[d2].v, false, kf.v, (short)0, s[j].v, false, false);
            }
        }

        // scale + causal mask
        const bool needmask = (k0 + 31) > qbase;
        #pragma unroll
        for (int j = 0; j < 2; j++)
            #pragma unroll
            for (int r = 0; r < 8; r++) {
                float v = s[j].f[r] * 0.125f;
                if (needmask) {
                    int qrow = qbase + r + 8 * hl;
                    int key  = k0 + j * 16 + l15;
                    if (key > qrow) v = -1e30f;
                }
                s[j].f[r] = v;
            }

        // online softmax (row spans 16 lanes of a wave half)
        #pragma unroll
        for (int r = 0; r < 8; r++) {
            float mx = fmaxf(s[0].f[r], s[1].f[r]);
            #pragma unroll
            for (int off = 8; off; off >>= 1) mx = fmaxf(mx, __shfl_xor(mx, off, 16));
            float mnew  = fmaxf(mrow[r], mx);
            float scale = __expf(mrow[r] - mnew);
            float p0 = __expf(s[0].f[r] - mnew);
            float p1 = __expf(s[1].f[r] - mnew);
            s[0].f[r] = p0; s[1].f[r] = p1;
            float rs = p0 + p1;
            #pragma unroll
            for (int off = 8; off; off >>= 1) rs += __shfl_xor(rs, off, 16);
            lrow[r] = lrow[r] * scale + rs;
            mrow[r] = mnew;
            #pragma unroll
            for (int t = 0; t < 4; t++) o[t].f[r] *= scale;
        }

        // P (C-layout f32) -> LDS -> A-layout bf16 fragment (per-wave buffer)
        unsigned short* pw = pb[wave];
        #pragma unroll
        for (int j = 0; j < 2; j++)
            #pragma unroll
            for (int r = 0; r < 8; r++) {
                int row = r + 8 * hl, col = j * 16 + l15;
                pw[row * 32 + col] = f2bf(s[j].f[r]);
            }
        wait_ds();
        FragBF pf;
        {
            int row = l15;
            #pragma unroll
            for (int i = 0; i < 8; i++) {
                int k = 2 * (i & 3) + ((i & 4) ? 16 : 0) + 8 * hl;
                pf.u[i] = *(const unsigned int*)&pw[row * 32 + k];
            }
        }
        // O += P @ V
        #pragma unroll
        for (int t = 0; t < 4; t++) {
            FragBF vf;
            int dcol = t * 16 + l15;
            #pragma unroll
            for (int i = 0; i < 8; i++) {
                int k = 2 * (i & 3) + ((i & 4) ? 16 : 0) + 8 * hl;
                vf.u[i] = *(const unsigned int*)&vtc[dcol * 32 + k];
            }
            o[t].v = __builtin_amdgcn_wmma_f32_16x16x32_bf16(
                false, pf.v, false, vf.v, (short)0, o[t].v, false, false);
        }
        // no trailing barrier: buffer reuse ordered by next wait_async+barrier
    }

    #pragma unroll
    for (int r = 0; r < 8; r++) {
        float inv = 1.0f / lrow[r];
        int qrow = qbase + r + 8 * hl;
        #pragma unroll
        for (int t = 0; t < 4; t++) {
            int col = h * HD + t * 16 + l15;
            Ob[(size_t)qrow * DIM + col] = f2bf(o[t].f[r] * inv);
        }
    }
}

// ---------------------------------------------------------------------------
extern "C" void kernel_launch(void* const* d_in, const int* in_sizes, int n_in,
                              void* d_out, int out_size, void* d_ws, size_t ws_size,
                              hipStream_t stream) {
    const float* x    = (const float*)d_in[0];
    const float* wq   = (const float*)d_in[1];
    const float* wk   = (const float*)d_in[2];
    const float* wv   = (const float*)d_in[3];
    const float* wo   = (const float*)d_in[4];
    const float* fcos = (const float*)d_in[5];
    const float* fsin = (const float*)d_in[6];
    // d_in[7] (mask) unused: causal mask applied analytically.
    float* out = (float*)d_out;

    char* p = (char*)d_ws;
    size_t off = 0;
    auto take = [&](size_t bytes) { char* r = p + off; off += (bytes + 255) & ~(size_t)255; return r; };
    unsigned short* xb   = (unsigned short*)take((size_t)SEQ * DIM * 2);
    unsigned short* wqT  = (unsigned short*)take((size_t)DIM * DIM * 2);   // [N][K]
    unsigned short* wkT  = (unsigned short*)take((size_t)KVD * DIM * 2);
    unsigned short* wvT  = (unsigned short*)take((size_t)KVD * DIM * 2);
    unsigned short* woT  = (unsigned short*)take((size_t)DIM * DIM * 2);
    float*          Qf   = (float*)take((size_t)SEQ * DIM * 4);
    float*          Kf   = (float*)take((size_t)SEQ * KVD * 4);
    float*          Vf   = (float*)take((size_t)SEQ * KVD * 4);
    unsigned short* Qhb  = (unsigned short*)take((size_t)NH  * SEQ * HD * 2);
    unsigned short* Khb  = (unsigned short*)take((size_t)NKV * SEQ * HD * 2);
    unsigned short* VhbT = (unsigned short*)take((size_t)NKV * HD * SEQ * 2);
    unsigned short* attb = (unsigned short*)take((size_t)SEQ * DIM * 2);
    (void)ws_size; (void)in_sizes; (void)n_in; (void)out_size;

    // 1. downconvert x; convert+transpose weights (one-time, keeps GEMM tiles
    //    async-copyable with no in-loop transposes)
    k_f32_to_bf16<<<2048, 256, 0, stream>>>(x, xb, SEQ * DIM);
    k_convT_bf16<<<(DIM * DIM + 255) / 256, 256, 0, stream>>>(wq, wqT, DIM);
    k_convT_bf16<<<(DIM * KVD + 255) / 256, 256, 0, stream>>>(wk, wkT, KVD);
    k_convT_bf16<<<(DIM * KVD + 255) / 256, 256, 0, stream>>>(wv, wvT, KVD);
    k_convT_bf16<<<(DIM * DIM + 255) / 256, 256, 0, stream>>>(wo, woT, DIM);

    // 2. QKV projections
    k_gemm_bf16<<<dim3(DIM / 256, SEQ / 64), 256, 0, stream>>>(xb, wqT, Qf, SEQ, DIM, DIM);
    k_gemm_bf16<<<dim3(KVD / 256, SEQ / 64), 256, 0, stream>>>(xb, wkT, Kf, SEQ, KVD, DIM);
    k_gemm_bf16<<<dim3(KVD / 256, SEQ / 64), 256, 0, stream>>>(xb, wvT, Vf, SEQ, KVD, DIM);

    // 3. RoPE + head-major repack (V transposed for async staging)
    k_rope_pack<<<(SEQ * NH  * HALF + 255) / 256, 256, 0, stream>>>(Qf, Qhb, fcos, fsin, NH);
    k_rope_pack<<<(SEQ * NKV * HALF + 255) / 256, 256, 0, stream>>>(Kf, Khb, fcos, fsin, NKV);
    k_v_packT<<<(SEQ * KVD + 255) / 256, 256, 0, stream>>>(Vf, VhbT);

    // 4. streaming-softmax attention
    k_flash_attn<<<dim3(SEQ / 64, NH), 128, 0, stream>>>(Qhb, Khb, VhbT, attb);

    // 5. output projection
    k_gemm_bf16<<<dim3(DIM / 256, SEQ / 64), 256, 0, stream>>>(attb, woT, out, SEQ, DIM, DIM);
}
